// FloodML_CNN_LSTM_82961588290363
// MI455X (gfx1250) — matrix operand order repaired and checked by reading, not compiled, not run on hardware
//
#include <hip/hip_runtime.h>

// ---------------------------------------------------------------------------
// FloodML CNN+LSTM for MI455X (gfx1250, wave32, WMMA + async LDS copies).
//
// Pipeline:
//   k1: f32->f16 weight conversion (w_ih, w_hh)
//   k2: fused conv1+relu+pool2 + conv2+relu+pool2 per frame (LDS-resident),
//       emits feats f16 (4320 x 6272)
//   k3: WMMA GEMM gates_x = feats @ w_ih^T + (b_ih+b_hh)   (4320 x 1024)
//       A chunks staged block-wide in LDS via global_load_async_to_lds_b128
//       (ASYNCcnt path), B streamed from L2-resident weights w/ prefetch.
//   k4: persistent single-workgroup LSTM (270 sequential steps, 16 waves,
//       h/c state in LDS, 32 WMMAs per wave per step) + fused head -> d_out
//
// Workspace layout (bytes), total ~82 MB:
//   feats_f16 : [0,            54,190,080)
//   w_ih_f16  : [54,190,080,   67,035,136)
//   w_hh_f16  : [67,035,136,   67,559,424)
//   gates_x   : [67,559,424,   85,254,144)
// ---------------------------------------------------------------------------

typedef _Float16 h8   __attribute__((ext_vector_type(8)));
typedef _Float16 v16h __attribute__((ext_vector_type(16)));
typedef float    v8f  __attribute__((ext_vector_type(8)));

union ABFrag { v16h v; h8 h[2]; };

#define BATCH 16
#define TSTEPS 270
#define BT 4320          // BATCH*TSTEPS
#define D_IN 6272
#define D_HID 256
#define G4 1024          // 4*D_HID
#define KCHUNK 448       // 6272 = 14 * 448; 14 k-steps of 32 per chunk

__device__ __forceinline__ float sigmoidf_(float x) {
  return 1.0f / (1.0f + expf(-x));
}

// Async 16B copy global -> LDS (per active lane). GVS addressing:
// mem = saddr + vaddr32; lds byte address from VDST vgpr. Tracked by ASYNCcnt.
__device__ __forceinline__ void async_copy_b128(uint32_t lds_off,
                                                const void* gbase,
                                                uint32_t voff) {
  asm volatile("global_load_async_to_lds_b128 %0, %1, %2"
               :: "v"(lds_off), "v"(voff), "s"(gbase)
               : "memory");
}
__device__ __forceinline__ void wait_asynccnt0() {
  asm volatile("s_wait_asynccnt 0x0" ::: "memory");
}

// ---------------------------------------------------------------- k1: f32->f16
__global__ void cvt_f16_kernel(const float* __restrict__ src,
                               _Float16* __restrict__ dst, int n) {
  int i = blockIdx.x * blockDim.x + threadIdx.x;
  int stride = gridDim.x * blockDim.x;
  for (; i < n; i += stride) dst[i] = (_Float16)src[i];
}

// --------------------------------------------- k2: fused conv stages per frame
// One 256-thread block per frame. LDS: img(15.4K f32) + f1(28.8K f16) +
// w2(9.2K f16) + small weights ~= 54.2 KB.
__global__ __launch_bounds__(256)
void conv_feat_kernel(const float* __restrict__ x,
                      const float* __restrict__ w1, const float* __restrict__ b1,
                      const float* __restrict__ w2, const float* __restrict__ b2,
                      _Float16* __restrict__ feats) {
  __shared__ __align__(16) float    s_img[62 * 62];
  __shared__ __align__(16) _Float16 s_f1[16 * 30 * 30];
  __shared__ __align__(16) _Float16 s_w2[32 * 16 * 9];
  __shared__ float s_w1[16 * 9];
  __shared__ float s_b1[16];
  __shared__ float s_b2[32];

  const int tid = threadIdx.x;
  const int f = blockIdx.x;
  const float* img = x + (size_t)f * 62 * 62;

  for (int i = tid; i < 3844; i += 256) s_img[i] = img[i];
  for (int i = tid; i < 144; i += 256) s_w1[i] = w1[i];
  for (int i = tid; i < 4608; i += 256) s_w2[i] = (_Float16)w2[i];
  if (tid < 16) s_b1[tid] = b1[tid];
  if (tid < 32) s_b2[tid] = b2[tid];
  __syncthreads();

  // stage 1: conv1(1->16, 3x3 valid) + relu + maxpool2 -> 16x30x30 (f16 LDS)
  for (int o = tid; o < 14400; o += 256) {
    const int ch = o / 900, rem = o % 900, py = rem / 30, px = rem % 30;
    const float* wk = s_w1 + ch * 9;
    const float bb = s_b1[ch];
    float mx = 0.0f;  // relu outputs are >= 0
    for (int dy = 0; dy < 2; ++dy)
      for (int dx = 0; dx < 2; ++dx) {
        const int iy = 2 * py + dy, ix = 2 * px + dx;
        float s = bb;
#pragma unroll
        for (int ky = 0; ky < 3; ++ky)
#pragma unroll
          for (int kx = 0; kx < 3; ++kx)
            s += wk[ky * 3 + kx] * s_img[(iy + ky) * 62 + ix + kx];
        mx = fmaxf(mx, fmaxf(s, 0.0f));
      }
    s_f1[o] = (_Float16)mx;
  }
  __syncthreads();

  // stage 2: conv2(16->32, 3x3 valid) + relu + maxpool2 -> 32x14x14 -> feats
  _Float16* out = feats + (size_t)f * D_IN;
  for (int o = tid; o < D_IN; o += 256) {
    const int co = o / 196, rem = o % 196, py = rem / 14, px = rem % 14;
    const _Float16* wc = s_w2 + co * 144;
    const float bb = s_b2[co];
    float mx = 0.0f;
    for (int dy = 0; dy < 2; ++dy)
      for (int dx = 0; dx < 2; ++dx) {
        const int iy = 2 * py + dy, ix = 2 * px + dx;
        float s = bb;
        for (int ci = 0; ci < 16; ++ci) {
          const _Float16* fp = s_f1 + ci * 900 + iy * 30 + ix;
          const _Float16* wp = wc + ci * 9;
#pragma unroll
          for (int ky = 0; ky < 3; ++ky)
#pragma unroll
            for (int kx = 0; kx < 3; ++kx)
              s += (float)wp[ky * 3 + kx] * (float)fp[ky * 30 + kx];
        }
        mx = fmaxf(mx, fmaxf(s, 0.0f));
      }
    out[o] = (_Float16)mx;
  }
}

// ------------------------------------- k3: gates_x = feats @ w_ih^T + bias
// Grid (2, 270), 256 threads = 8 waves. Wave handles a 16(M) x 64(N) strip.
// The A tile (16 rows) is identical for all 8 waves -> stage each 16xKCHUNK
// chunk once per block in LDS with async copies, then ds_load fragments.
// A-frag (16x32 f16): lane L: m=L%16, g=L/16; halves [k0+8g, +8) and
//                     [k0+16+8g, +8) of A row m.
// B-frag (32x16 f16): lane L: n=L%16, kk=L/16; halves [k0+8kk, +8) and
//                     [k0+16+8kk, +8) of w_ih[n] (w_ih row-major == B^T row-major).
// D (16x16 f32):      lane L: n=L%16, vgpr r -> m = r + 8*(L/16).
__global__ __launch_bounds__(256)
void gemm_x_kernel(const _Float16* __restrict__ feats,
                   const _Float16* __restrict__ wih,
                   const float* __restrict__ b_ih,
                   const float* __restrict__ b_hh,
                   float* __restrict__ gatesx) {
  __shared__ __align__(16) _Float16 s_a[16 * KCHUNK];  // 14 KB

  const int tid = threadIdx.x;
  const int lane = tid & 31;
  const int wave = tid >> 5;
  const int mBase = blockIdx.y * 16;
  const int nBase = blockIdx.x * 512 + wave * 64;
  const int lm = lane & 15, g = lane >> 4;

  const _Float16* brow[4];
#pragma unroll
  for (int t = 0; t < 4; ++t)
    brow[t] = wih + (size_t)(nBase + t * 16 + lm) * D_IN + g * 8;

  v8f acc[4] = {};
  const _Float16* arow_lds = s_a + lm * KCHUNK;

  for (int kc = 0; kc < D_IN; kc += KCHUNK) {
    __syncthreads();  // previous chunk fully consumed
    // cooperative async copy: 16 x KCHUNK halves = 896 x 16B transfers
    for (int j = tid; j < 16 * KCHUNK / 8; j += 256) {
      const int row = j / (KCHUNK / 8);
      const int c16 = j % (KCHUNK / 8);
      const uint32_t lds_off =
          (uint32_t)(uintptr_t)(s_a + row * KCHUNK + c16 * 8);
      const uint32_t voff =
          (uint32_t)(((size_t)(mBase + row) * D_IN + kc + c16 * 8) *
                     sizeof(_Float16));
      async_copy_b128(lds_off, feats, voff);
    }
    // prefetch next chunk of this wave's B rows into L2/L0
    if (kc + KCHUNK < D_IN) {
#pragma unroll
      for (int t = 0; t < 4; ++t)
        __builtin_prefetch(brow[t] + kc + KCHUNK, 0, 0);
    }
    wait_asynccnt0();
    __syncthreads();  // staged A visible to all waves

    for (int k0 = 0; k0 < KCHUNK; k0 += 32) {
      ABFrag a;
      a.h[0] = *(const h8*)(arow_lds + k0 + 8 * g);
      a.h[1] = *(const h8*)(arow_lds + k0 + 16 + 8 * g);
#pragma unroll
      for (int t = 0; t < 4; ++t) {
        ABFrag b;
        b.h[0] = *(const h8*)(brow[t] + kc + k0);
        b.h[1] = *(const h8*)(brow[t] + kc + k0 + 16);
        acc[t] = __builtin_amdgcn_wmma_f32_16x16x32_f16(
            false, a.v, false, b.v, (short)0, acc[t], false, false);
      }
    }
  }

#pragma unroll
  for (int t = 0; t < 4; ++t) {
    const int n = nBase + t * 16 + lm;
    const float bias = b_ih[n] + b_hh[n];
#pragma unroll
    for (int r = 0; r < 8; ++r) {
      const int m = mBase + r + 8 * g;
      gatesx[(size_t)m * G4 + n] = acc[t][r] + bias;
    }
  }
}

// --------------------------- k4: persistent LSTM recurrence + head (1 block)
// 512 threads = 16 waves; wave w owns N-range [w*64, w*64+64).
// LDS: h f16 (8K) + gates f16 (32K) + c f32 (16K) = 56 KB.
__global__ __launch_bounds__(512)
void lstm_head_kernel(const float* __restrict__ gatesx,
                      const _Float16* __restrict__ whh,
                      const float* __restrict__ h0,
                      const float* __restrict__ c0,
                      const float* __restrict__ head_w,
                      const float* __restrict__ head_b,
                      float* __restrict__ out) {
  __shared__ __align__(16) _Float16 s_h[BATCH * D_HID];
  __shared__ __align__(16) _Float16 s_g[BATCH * G4];
  __shared__ float s_c[BATCH * D_HID];

  const int tid = threadIdx.x;
  const int lane = tid & 31;
  const int wave = tid >> 5;
  const int lm = lane & 15, g = lane >> 4;

  for (int i = tid; i < BATCH * D_HID; i += 512) {
    s_h[i] = (_Float16)h0[i];
    s_c[i] = c0[i];
  }
  __syncthreads();

  const _Float16* hrow = s_h + lm * D_HID;
  const _Float16* brow[4];
#pragma unroll
  for (int t = 0; t < 4; ++t)
    brow[t] = whh + (size_t)(wave * 64 + t * 16 + lm) * D_HID + g * 8;

  for (int ts = 0; ts < TSTEPS; ++ts) {
    // init accumulators from precomputed x-projection (bias already folded)
    v8f acc[4];
#pragma unroll
    for (int t = 0; t < 4; ++t) {
      const int n = wave * 64 + t * 16 + lm;
#pragma unroll
      for (int r = 0; r < 8; ++r) {
        const int m = r + 8 * g;  // batch index
        acc[t][r] = gatesx[(size_t)(m * TSTEPS + ts) * G4 + n];
      }
    }
    // gates += h @ w_hh^T
    for (int k0 = 0; k0 < D_HID; k0 += 32) {
      ABFrag a;
      a.h[0] = *(const h8*)(hrow + k0 + 8 * g);
      a.h[1] = *(const h8*)(hrow + k0 + 16 + 8 * g);
#pragma unroll
      for (int t = 0; t < 4; ++t) {
        ABFrag b;
        b.h[0] = *(const h8*)(brow[t] + k0);
        b.h[1] = *(const h8*)(brow[t] + k0 + 16);
        acc[t] = __builtin_amdgcn_wmma_f32_16x16x32_f16(
            false, a.v, false, b.v, (short)0, acc[t], false, false);
      }
    }
    // stage gates in LDS
#pragma unroll
    for (int t = 0; t < 4; ++t) {
      const int n = wave * 64 + t * 16 + lm;
#pragma unroll
      for (int r = 0; r < 8; ++r) {
        const int m = r + 8 * g;
        s_g[m * G4 + n] = (_Float16)acc[t][r];
      }
    }
    __syncthreads();
    // elementwise LSTM cell update
    for (int i = tid; i < BATCH * D_HID; i += 512) {
      const int m = i >> 8, n = i & 255;
      const float gi = sigmoidf_((float)s_g[m * G4 + n]);
      const float gf = sigmoidf_((float)s_g[m * G4 + D_HID + n]);
      const float gg = tanhf((float)s_g[m * G4 + 2 * D_HID + n]);
      const float go = sigmoidf_((float)s_g[m * G4 + 3 * D_HID + n]);
      const float c = gf * s_c[i] + gi * gg;
      s_c[i] = c;
      s_h[i] = (_Float16)(go * tanhf(c));
    }
    __syncthreads();
  }

  // head: out[b] = h_last[b] . head_w + head_b
  if (tid < BATCH) {
    float s = head_b[0];
    for (int n = 0; n < D_HID; ++n)
      s += (float)s_h[tid * D_HID + n] * head_w[n];
    out[tid] = s;
  }
}

// ---------------------------------------------------------------------------
extern "C" void kernel_launch(void* const* d_in, const int* in_sizes, int n_in,
                              void* d_out, int out_size, void* d_ws,
                              size_t ws_size, hipStream_t stream) {
  const float* x       = (const float*)d_in[0];
  const float* conv1_w = (const float*)d_in[1];
  const float* conv1_b = (const float*)d_in[2];
  const float* conv2_w = (const float*)d_in[3];
  const float* conv2_b = (const float*)d_in[4];
  const float* w_ih    = (const float*)d_in[5];
  const float* w_hh    = (const float*)d_in[6];
  const float* b_ih    = (const float*)d_in[7];
  const float* b_hh    = (const float*)d_in[8];
  const float* head_w  = (const float*)d_in[9];
  const float* head_b  = (const float*)d_in[10];
  const float* h0      = (const float*)d_in[11];
  const float* c0      = (const float*)d_in[12];

  char* ws = (char*)d_ws;
  const size_t FEATS_B = (size_t)BT * D_IN * sizeof(_Float16);   // 54,190,080
  const size_t WIH_B   = (size_t)G4 * D_IN * sizeof(_Float16);   // 12,845,056
  const size_t WHH_B   = (size_t)G4 * D_HID * sizeof(_Float16);  //    524,288
  _Float16* feats  = (_Float16*)(ws);
  _Float16* wih16  = (_Float16*)(ws + FEATS_B);
  _Float16* whh16  = (_Float16*)(ws + FEATS_B + WIH_B);
  float*    gatesx = (float*)(ws + FEATS_B + WIH_B + WHH_B);

  // k1: weight conversions
  cvt_f16_kernel<<<4096, 256, 0, stream>>>(w_ih, wih16, G4 * D_IN);
  cvt_f16_kernel<<<1024, 256, 0, stream>>>(w_hh, whh16, G4 * D_HID);

  // k2: CNN features (one block per frame)
  conv_feat_kernel<<<BT, 256, 0, stream>>>(x, conv1_w, conv1_b, conv2_w,
                                           conv2_b, feats);

  // k3: input projection GEMM with WMMA + async LDS staging of A
  gemm_x_kernel<<<dim3(2, 270), 256, 0, stream>>>(feats, wih16, b_ih, b_hh,
                                                  gatesx);

  // k4: sequential LSTM + head (single persistent workgroup)
  lstm_head_kernel<<<1, 512, 0, stream>>>(gatesx, whh16, h0, c0, head_w,
                                          head_b, (float*)d_out);
}
